// RGN_72962904424634
// MI455X (gfx1250) — matrix operand order repaired and checked by reading, not compiled, not run on hardware
//
#include <hip/hip_runtime.h>
#include <hip/hip_bf16.h>

typedef __bf16 bf16_t;
typedef bf16_t v16bf __attribute__((ext_vector_type(16)));
typedef float  v8f   __attribute__((ext_vector_type(8)));

#define L_SEQ 512
#define BATCH 16
#define D_IN  41
#define HID   800
#define ALPHA 20
#define NT    (4 * HID / 16)   // 200 gate N-tiles of 16 cols
#define KH_CH (HID / 32)       // 25 K-chunks for recurrent part
#define KX0_CH 2               // ceil(41/32) K-chunks for layer-0 input
#define KX1_CH (2 * HID / 32)  // 50 K-chunks for layer-1 input
#define NWG   (HID / 16)       // 50 persistent workgroups (one hidden tile each)

#define WMMA_BF16(A, B, C) \
    __builtin_amdgcn_wmma_f32_16x16x32_bf16(false, (A), false, (B), (short)0, (C), false, false)

// ---------------------------------------------------------------------------
// WMMA fragment packing (CDNA5 ISA 7.12.2, wave32)
// A (16x32 bf16, M=batch, K): per-lane 16 elems:
//   kl<8  -> lane=m,    e=kl
//   kl<16 -> lane=m+16, e=kl-8
//   kl<24 -> lane=m,    e=kl-8
//   else  -> lane=m+16, e=kl-16
// B (32x16 bf16, K x N): same k split as A, lane = n or n+16
// Each fragment block = 32 lanes * 16 bf16 = 512 elems = 1 KB, lane-contiguous.
// ---------------------------------------------------------------------------

__device__ __forceinline__ void store_apack(bf16_t* base, int m, int k, float v) {
    int kc = k >> 5, kl = k & 31;
    int lane, e;
    if (kl < 8)       { lane = m;      e = kl;      }
    else if (kl < 16) { lane = m + 16; e = kl - 8;  }
    else if (kl < 24) { lane = m;      e = kl - 8;  }
    else              { lane = m + 16; e = kl - 16; }
    base[(size_t)kc * 512 + lane * 16 + e] = (bf16_t)v;
}

// Async DMA: 16 bytes global -> LDS per lane, tracked by ASYNCcnt (ISA 15.18.3 op 98).
__device__ __forceinline__ void async_ld_b128(unsigned lds_off, const void* gsrc) {
    asm volatile("global_load_async_to_lds_b128 %0, %1, off"
                 :: "v"(lds_off), "v"((unsigned long long)(size_t)gsrc)
                 : "memory");
}
__device__ __forceinline__ void wait_asynccnt0() {
    asm volatile("s_wait_asynccnt 0x0" ::: "memory");
}

// Pack weight rows [Nrows=3200 x K] fp32 row-major into B-fragment blocks.
__global__ void pack_B_kernel(const float* __restrict__ W, int K, int kc_base,
                              int kc_count, int kch_total, bf16_t* __restrict__ out) {
    size_t idx = (size_t)blockIdx.x * blockDim.x + threadIdx.x;
    size_t total = (size_t)NT * kc_count * 512;
    if (idx >= total) return;
    int e    = idx & 15;
    int lane = (idx >> 4) & 31;
    int kc   = (int)((idx >> 9) % kc_count);
    int n    = (int)(idx / ((size_t)kc_count * 512));
    int j    = lane & 15;
    int kl   = ((e < 8) ? e : (e + 8)) + ((lane < 16) ? 0 : 8);
    int k    = kc * 32 + kl;
    int row  = n * 16 + j;
    float v  = (k < K) ? W[(size_t)row * K + k] : 0.0f;
    out[((size_t)n * kch_total + kc_base + kc) * 512 + lane * 16 + e] = (bf16_t)v;
}

// Pack x [BATCH, L, D_IN] fp32 into per-timestep A-fragment blocks, K padded to 64.
__global__ void pack_x0_kernel(const float* __restrict__ x, bf16_t* __restrict__ out) {
    size_t idx = (size_t)blockIdx.x * blockDim.x + threadIdx.x;
    size_t total = (size_t)L_SEQ * KX0_CH * 512;
    if (idx >= total) return;
    int e    = idx & 15;
    int lane = (idx >> 4) & 31;
    int kc   = (int)((idx >> 9) % KX0_CH);
    int t    = (int)(idx / ((size_t)KX0_CH * 512));
    int m, kl;
    if (lane < 16) { m = lane;      kl = (e < 8) ? e       : e + 8;  }
    else           { m = lane - 16; kl = (e < 8) ? (e + 8) : e + 16; }
    int k = kc * 32 + kl;
    float v = (k < D_IN) ? x[((size_t)m * L_SEQ + t) * D_IN + k] : 0.0f;
    out[idx] = (bf16_t)v;
}

__global__ void bias_sum_kernel(const float* __restrict__ bi, const float* __restrict__ bh,
                                float* __restrict__ out) {
    int i = blockIdx.x * blockDim.x + threadIdx.x;
    if (i < 4 * HID) out[i] = bi[i] + bh[i];
}

__global__ void init_counters_kernel(unsigned* bar) {
    if (threadIdx.x < 8) bar[threadIdx.x] = 0u;
}

__device__ __forceinline__ float sigmoidf_(float x) { return 1.0f / (1.0f + __expf(-x)); }

__device__ __forceinline__ void grid_barrier(unsigned* cnt, unsigned target) {
    __syncthreads();
    if (threadIdx.x == 0) {
        __threadfence();
        atomicAdd(cnt, 1u);
        volatile unsigned* vc = (volatile unsigned*)cnt;
        while (*vc < target) { __builtin_amdgcn_s_sleep(2); }
    }
    __syncthreads();
    __threadfence();
}

// ---------------------------------------------------------------------------
// Persistent LSTM direction kernel.
// Grid = NWG (one WG per 16-hidden-unit tile), block = 256 (8 waves).
// wave = gate(0..3) x K-half(0..1). Per step:
//   gates[16 x 3200] = [h_prev | x_t](bf16, A-packed) @ Bpack + bias
// A fragments staged via global_load_async_to_lds_b128. The K-loop is a
// branch-free software pipeline: 2 accumulators, fragments always written by
// loads (never VALU moves) to stay out of the WMMA->VALU hazard window.
// ---------------------------------------------------------------------------
__global__ void __launch_bounds__(256)
lstm_persist_kernel(const bf16_t* __restrict__ Bpack, int kch_total,
                    const bf16_t* __restrict__ xpack, int kx_ch,
                    const float* __restrict__ bias,
                    bf16_t* __restrict__ hpack,       // ping-pong [2][KH_CH*512]
                    bf16_t* __restrict__ dest_pack,   // A-packed dest (layer0 -> layer1), or null
                    int dest_col_base, int dest_kch_total,
                    float* __restrict__ dest_rm,      // row-major dest (layer1 -> head), or null
                    int dest_rm_cols, int dest_rm_colbase,
                    int reverse, unsigned* bar) {
    __shared__ float gpart[8][16][16];
    __shared__ float cst[16][16];
    extern __shared__ bf16_t Ash[];   // (KH_CH + kx_ch) fragment blocks

    const int ht   = blockIdx.x;      // hidden tile 0..49
    const int tid  = threadIdx.x;
    const int wave = tid >> 5;
    const int lane = tid & 31;
    const int gate = wave & 3;
    const int half = wave >> 2;
    const int ntile = gate * (HID / 16) + ht;      // 0..199
    const int hc = (kch_total + 1) >> 1;
    const int c0 = half ? hc : 0;
    const int c1 = half ? kch_total : hc;

    // Zero h0 cooperatively (2*KH_CH*512 bf16 = 12800 uints over NWG*256 threads).
    unsigned* hz = (unsigned*)hpack;
    for (int i = blockIdx.x * blockDim.x + tid; i < KH_CH * 512; i += NWG * 256)
        hz[i] = 0u;
    cst[tid >> 4][tid & 15] = 0.0f;
    __threadfence();
    grid_barrier(bar, (unsigned)NWG * 1u);

    const unsigned AshOff = (unsigned)(size_t)(void*)Ash;  // generic->LDS offset (low 32b)
    const bf16_t* Bt = Bpack + (size_t)ntile * kch_total * 512;
    int pp = 0;
    for (int s = 0; s < L_SEQ; ++s) {
        const int t = reverse ? (L_SEQ - 1 - s) : s;

        // Async-DMA the A fragments (h_prev, then x_t) into LDS: 16B per lane.
        {
            const char* srcH = (const char*)(hpack + (size_t)pp * KH_CH * 512);
            const int nH = KH_CH * 64;               // 16B units for h part
            for (int i = tid; i < nH; i += 256)
                async_ld_b128(AshOff + (unsigned)i * 16u, srcH + (size_t)i * 16);
            const char* srcX = (const char*)(xpack + (size_t)t * kx_ch * 512);
            const int nX = kx_ch * 64;
            for (int i = tid; i < nX; i += 256)
                async_ld_b128(AshOff + (unsigned)(nH + i) * 16u, srcX + (size_t)i * 16);
            wait_asynccnt0();
        }
        // Warm next timestep's x fragments into L2/WGP$ while we compute.
        {
            const int tn = reverse ? (t > 0 ? t - 1 : 0) : (t + 1 < L_SEQ ? t + 1 : t);
            if (tid < kx_ch)
                __builtin_prefetch((const void*)(xpack + ((size_t)tn * kx_ch + tid) * 512), 0, 1);
        }
        __syncthreads();

        // Gate GEMM over K-chunks [c0, c1): branch-free pipelined loop.
        v8f acc0 = {}, acc1 = {};
        {
            const bf16_t* Ap = Ash + (size_t)c0 * 512 + lane * 16;
            const bf16_t* Bp = Bt + (size_t)c0 * 512 + lane * 16;
            const int nk = c1 - c0;
            int kc = 0;
            if (nk >= 2) {
                v16bf a0 = *(const v16bf*)(Ap);
                v16bf B0 = *(const v16bf*)(Bp);
                v16bf a1 = *(const v16bf*)(Ap + 512);
                v16bf B1 = *(const v16bf*)(Bp + 512);
                #pragma unroll 2
                for (kc = 0; kc + 3 < nk; kc += 2) {
                    v16bf a2 = *(const v16bf*)(Ap + (size_t)(kc + 2) * 512);
                    v16bf B2 = *(const v16bf*)(Bp + (size_t)(kc + 2) * 512);
                    v16bf a3 = *(const v16bf*)(Ap + (size_t)(kc + 3) * 512);
                    v16bf B3 = *(const v16bf*)(Bp + (size_t)(kc + 3) * 512);
                    acc0 = WMMA_BF16(a0, B0, acc0);
                    acc1 = WMMA_BF16(a1, B1, acc1);
                    a0 = a2; B0 = B2; a1 = a3; B1 = B3;
                }
                acc0 = WMMA_BF16(a0, B0, acc0);
                acc1 = WMMA_BF16(a1, B1, acc1);
                kc += 2;
            }
            for (; kc < nk; ++kc) {   // odd tail (at most one chunk)
                v16bf a = *(const v16bf*)(Ap + (size_t)kc * 512);
                v16bf b = *(const v16bf*)(Bp + (size_t)kc * 512);
                acc0 = WMMA_BF16(a, b, acc0);
            }
        }
        const int mofs = (lane >= 16) ? 8 : 0;
        #pragma unroll
        for (int v = 0; v < 8; ++v)
            gpart[wave][v + mofs][lane & 15] = acc0[v] + acc1[v];
        __syncthreads();

        // Pointwise cell update: one (batch, j) pair per thread.
        {
            const int b = tid >> 4, j = tid & 15;
            const int col = ht * 16 + j;
            const float gi = gpart[0][b][j] + gpart[4][b][j] + bias[0 * HID + col];
            const float gf = gpart[1][b][j] + gpart[5][b][j] + bias[1 * HID + col];
            const float gg = gpart[2][b][j] + gpart[6][b][j] + bias[2 * HID + col];
            const float go = gpart[3][b][j] + gpart[7][b][j] + bias[3 * HID + col];
            const float ig = sigmoidf_(gi);
            const float fg = sigmoidf_(gf);
            const float gt = tanhf(gg);
            const float og = sigmoidf_(go);
            const float c  = fg * cst[b][j] + ig * gt;
            cst[b][j] = c;
            const float h = og * tanhf(c);
            store_apack(hpack + (size_t)(pp ^ 1) * KH_CH * 512, b, col, h);
            if (dest_pack)
                store_apack(dest_pack + (size_t)t * dest_kch_total * 512, b,
                            dest_col_base + col, h);
            else
                dest_rm[((size_t)t * BATCH + b) * dest_rm_cols + dest_rm_colbase + col] = h;
        }
        __threadfence();
        grid_barrier(bar, (unsigned)NWG * (unsigned)(s + 2));
        pp ^= 1;
    }
}

// logits[l,b,a] = h2[l,b,:] . W_lin[a,:] + b_lin[a]
__global__ void logits_kernel(const float* __restrict__ h2, const float* __restrict__ W,
                              const float* __restrict__ bl, float* __restrict__ logits) {
    int idx = blockIdx.x * blockDim.x + threadIdx.x;
    if (idx >= L_SEQ * BATCH * ALPHA) return;
    const int a = idx % ALPHA;
    const int lb = idx / ALPHA;
    const float* hrow = h2 + (size_t)lb * (2 * HID);
    const float* wrow = W + (size_t)a * (2 * HID);
    float s = bl[a];
    for (int k = 0; k < 2 * HID; ++k) s += hrow[k] * wrow[k];
    logits[idx] = s;
}

// softmax over BATCH axis (faithful nn.Softmax(dim=1)) then dihedral mixture.
__global__ void softdih_kernel(const float* __restrict__ logits,
                               const float* __restrict__ alphabet,
                               float* __restrict__ dih) {
    const int l = blockIdx.x;
    __shared__ float lg[BATCH][ALPHA];
    __shared__ float pm[BATCH][ALPHA];
    __shared__ float mx[ALPHA], sm[ALPHA];
    const int tid = threadIdx.x;
    for (int i = tid; i < BATCH * ALPHA; i += 64)
        lg[i / ALPHA][i % ALPHA] = logits[(size_t)l * BATCH * ALPHA + i];
    __syncthreads();
    if (tid < ALPHA) {
        float m = -1e30f;
        for (int b = 0; b < BATCH; ++b) m = fmaxf(m, lg[b][tid]);
        float s = 0.0f;
        for (int b = 0; b < BATCH; ++b) s += __expf(lg[b][tid] - m);
        mx[tid] = m; sm[tid] = s;
    }
    __syncthreads();
    for (int i = tid; i < BATCH * ALPHA; i += 64) {
        const int b = i / ALPHA, a = i % ALPHA;
        pm[b][a] = __expf(lg[b][a] - mx[a]) / sm[a];
    }
    __syncthreads();
    if (tid < BATCH * 3) {
        const int b = tid / 3, j = tid % 3;
        float ssum = 0.0f, csum = 0.0f;
        for (int a = 0; a < ALPHA; ++a) {
            const float ang = alphabet[a * 3 + j];
            ssum += pm[b][a] * __sinf(ang);
            csum += pm[b][a] * __cosf(ang);
        }
        dih[((size_t)l * BATCH + b) * 3 + j] = atan2f(ssum, csum);
    }
}

// Sequential NeRF backbone extension: 3L steps, one lane per batch element.
__global__ void nerf_kernel(const float* __restrict__ dih, float* __restrict__ out) {
    const int b = threadIdx.x;
    if (b >= BATCH) return;
    const float BL[3] = {145.801f, 152.326f, 132.868f};
    const float BA[3] = {2.124f, 1.941f, 2.028f};
    float ax = 0.f, ay = 0.f, az = 0.f;
    float bx = BL[0], by = 0.f, bz = 0.f;
    const float ang = 3.14159265358979f - BA[0];
    float cx = bx + BL[1] * __cosf(ang), cy = BL[1] * __sinf(ang), cz = 0.f;
    for (int i = 0; i < 3 * L_SEQ; ++i) {
        const int l = i / 3, j = i % 3;
        const float r = BL[j], th = BA[j];
        const float ph = dih[((size_t)l * BATCH + b) * 3 + j];
        const float st = __sinf(th), ct = __cosf(th);
        const float d2x = -r * ct, d2y = r * st * __cosf(ph), d2z = r * st * __sinf(ph);
        float bcx = cx - bx, bcy = cy - by, bcz = cz - bz;
        float inv = 1.0f / (sqrtf(bcx * bcx + bcy * bcy + bcz * bcz) + 1e-8f);
        bcx *= inv; bcy *= inv; bcz *= inv;
        const float abx = bx - ax, aby = by - ay, abz = bz - az;
        float nx = aby * bcz - abz * bcy;
        float ny = abz * bcx - abx * bcz;
        float nz = abx * bcy - aby * bcx;
        inv = 1.0f / (sqrtf(nx * nx + ny * ny + nz * nz) + 1e-8f);
        nx *= inv; ny *= inv; nz *= inv;
        const float mx_ = ny * bcz - nz * bcy;
        const float my_ = nz * bcx - nx * bcz;
        const float mz_ = nx * bcy - ny * bcx;
        const float dx = bcx * d2x + mx_ * d2y + nx * d2z + cx;
        const float dy = bcy * d2x + my_ * d2y + ny * d2z + cy;
        const float dz = bcz * d2x + mz_ * d2y + nz * d2z + cz;
        out[((size_t)i * BATCH + b) * 3 + 0] = dx;
        out[((size_t)i * BATCH + b) * 3 + 1] = dy;
        out[((size_t)i * BATCH + b) * 3 + 2] = dz;
        ax = bx; ay = by; az = bz;
        bx = cx; by = cy; bz = cz;
        cx = dx; cy = dy; cz = dz;
    }
}

extern "C" void kernel_launch(void* const* d_in, const int* in_sizes, int n_in,
                              void* d_out, int out_size, void* d_ws, size_t ws_size,
                              hipStream_t stream) {
    (void)in_sizes; (void)n_in; (void)out_size; (void)ws_size;
    const float* x = (const float*)d_in[0];
    const float *Wih[4], *Whh[4], *bih[4], *bhh[4];
    for (int d = 0; d < 4; ++d) {
        Wih[d] = (const float*)d_in[1 + 4 * d];
        Whh[d] = (const float*)d_in[2 + 4 * d];
        bih[d] = (const float*)d_in[3 + 4 * d];
        bhh[d] = (const float*)d_in[4 + 4 * d];
    }
    const float* W_lin   = (const float*)d_in[17];
    const float* b_lin   = (const float*)d_in[18];
    const float* alphab  = (const float*)d_in[19];
    float* out = (float*)d_out;

    const int KCH0 = KH_CH + KX0_CH;   // 27
    const int KCH1 = KH_CH + KX1_CH;   // 75

    // Workspace layout
    char* ws = (char*)d_ws;
    size_t off = 0;
    auto alloc = [&](size_t bytes) { size_t o = off; off += (bytes + 255) & ~(size_t)255; return o; };
    const size_t o_bar    = alloc(256);
    const size_t o_bias   = alloc((size_t)4 * 4 * HID * sizeof(float));
    size_t o_bpack[4];
    o_bpack[0] = alloc((size_t)NT * KCH0 * 512 * sizeof(bf16_t));
    o_bpack[1] = alloc((size_t)NT * KCH0 * 512 * sizeof(bf16_t));
    o_bpack[2] = alloc((size_t)NT * KCH1 * 512 * sizeof(bf16_t));
    o_bpack[3] = alloc((size_t)NT * KCH1 * 512 * sizeof(bf16_t));
    const size_t o_xpack0 = alloc((size_t)L_SEQ * KX0_CH * 512 * sizeof(bf16_t));
    const size_t o_h1pack = alloc((size_t)L_SEQ * KX1_CH * 512 * sizeof(bf16_t));
    const size_t o_h2     = alloc((size_t)L_SEQ * BATCH * 2 * HID * sizeof(float));
    const size_t o_hpack  = alloc((size_t)2 * KH_CH * 512 * sizeof(bf16_t));
    const size_t o_logits = alloc((size_t)L_SEQ * BATCH * ALPHA * sizeof(float));
    const size_t o_dih    = alloc((size_t)L_SEQ * BATCH * 3 * sizeof(float));

    unsigned* bar   = (unsigned*)(ws + o_bar);
    float* biasbuf  = (float*)(ws + o_bias);
    bf16_t* bpk[4]; for (int d = 0; d < 4; ++d) bpk[d] = (bf16_t*)(ws + o_bpack[d]);
    bf16_t* xpack0  = (bf16_t*)(ws + o_xpack0);
    bf16_t* h1pack  = (bf16_t*)(ws + o_h1pack);
    float* h2       = (float*)(ws + o_h2);
    bf16_t* hpack   = (bf16_t*)(ws + o_hpack);
    float* logits   = (float*)(ws + o_logits);
    float* dih      = (float*)(ws + o_dih);

    init_counters_kernel<<<1, 32, 0, stream>>>(bar);

    // Pack weights into B-fragment blocks: [W_hh | W_ih] along the K-chunk axis.
    for (int d = 0; d < 4; ++d) {
        const int kcht = (d < 2) ? KCH0 : KCH1;
        const int kxc  = (d < 2) ? KX0_CH : KX1_CH;
        const int Kin  = (d < 2) ? D_IN : 2 * HID;
        {
            size_t tot = (size_t)NT * KH_CH * 512;
            pack_B_kernel<<<(unsigned)((tot + 255) / 256), 256, 0, stream>>>(
                Whh[d], HID, 0, KH_CH, kcht, bpk[d]);
        }
        {
            size_t tot = (size_t)NT * kxc * 512;
            pack_B_kernel<<<(unsigned)((tot + 255) / 256), 256, 0, stream>>>(
                Wih[d], Kin, KH_CH, kxc, kcht, bpk[d]);
        }
        bias_sum_kernel<<<(4 * HID + 255) / 256, 256, 0, stream>>>(
            bih[d], bhh[d], biasbuf + (size_t)d * 4 * HID);
    }
    {
        size_t tot = (size_t)L_SEQ * KX0_CH * 512;
        pack_x0_kernel<<<(unsigned)((tot + 255) / 256), 256, 0, stream>>>(x, xpack0);
    }

    const size_t sh0 = (size_t)KCH0 * 512 * sizeof(bf16_t);   // 27 KB A staging
    const size_t sh1 = (size_t)KCH1 * 512 * sizeof(bf16_t);   // 75 KB A staging

    // Layer 0 forward / backward -> h1pack (A-packed layer-1 input, cols [0,800) / [800,1600))
    lstm_persist_kernel<<<NWG, 256, sh0, stream>>>(
        bpk[0], KCH0, xpack0, KX0_CH, biasbuf + 0 * 4 * HID,
        hpack, h1pack, 0, KX1_CH, nullptr, 0, 0, 0, bar + 0);
    lstm_persist_kernel<<<NWG, 256, sh0, stream>>>(
        bpk[1], KCH0, xpack0, KX0_CH, biasbuf + 1 * 4 * HID,
        hpack, h1pack, HID, KX1_CH, nullptr, 0, 0, 1, bar + 1);
    // Layer 1 forward / backward -> h2 row-major [L, B, 1600]
    lstm_persist_kernel<<<NWG, 256, sh1, stream>>>(
        bpk[2], KCH1, h1pack, KX1_CH, biasbuf + 2 * 4 * HID,
        hpack, nullptr, 0, 0, h2, 2 * HID, 0, 0, bar + 2);
    lstm_persist_kernel<<<NWG, 256, sh1, stream>>>(
        bpk[3], KCH1, h1pack, KX1_CH, biasbuf + 3 * 4 * HID,
        hpack, nullptr, 0, 0, h2, 2 * HID, HID, 1, bar + 3);

    logits_kernel<<<(L_SEQ * BATCH * ALPHA + 255) / 256, 256, 0, stream>>>(
        h2, W_lin, b_lin, logits);
    softdih_kernel<<<L_SEQ, 64, 0, stream>>>(logits, alphab, dih);
    nerf_kernel<<<1, 32, 0, stream>>>(dih, out);
}
